// FourierGPT_9844065042504
// MI455X (gfx1250) — matrix operand order repaired and checked by reading, not compiled, not run on hardware
//
#include <hip/hip_runtime.h>
#include <hip/hip_bf16.h>

// FourierGPT forward on gfx1250 (MI455X).
// real(fft(x)) over channels == x @ cos-DFT matrix, so every heavy op is a
// bf16 WMMA GEMM (V_WMMA_F32_16X16X32_BF16, f32 accumulate).
// GEMM: 8 waves/block, 32x64 tile/wave, double-buffered K-loop with
// unconditional in-loop prefetch (partial loadcnt waits), persistent per-lane
// pointers, and explicit global-address-space loads (global_load_b128, not
// flat_load_b128, so LOADcnt is decoupled from DScnt).

typedef __attribute__((ext_vector_type(16))) __bf16       v16bf;
typedef __attribute__((ext_vector_type(8)))  float        v8f;
typedef __attribute__((ext_vector_type(4)))  unsigned int u32x4;  // native, not HIP class

// Global-address-space vector (forces global_load_b128).
typedef __attribute__((address_space(1))) const u32x4 GQ;

#define DIM   1024
#define SEQ   2048
#define NTOK  4096      // B*T = 2*2048
#define HID   4096      // 4*DIM
#define VOCAB 32000
#define DEPTH 12
#define LN_EPS 1e-5f

// ---------------------------------------------------------------- helpers
__device__ __forceinline__ float gelu_exact(float x) {
    return 0.5f * x * (1.0f + erff(x * 0.70710678118654752f));
}

// Row statistics over one 1024-float row, block of 256 threads, 4 elems/thread.
__device__ __forceinline__ void row_stats(float4 x, float& mean, float& rstd) {
    float s  = x.x + x.y + x.z + x.w;
    float s2 = x.x*x.x + x.y*x.y + x.z*x.z + x.w*x.w;
    #pragma unroll
    for (int off = 16; off > 0; off >>= 1) {          // wave32 reduction
        s  += __shfl_xor(s,  off, 32);
        s2 += __shfl_xor(s2, off, 32);
    }
    __shared__ float red[16];
    const int w = threadIdx.x >> 5;
    if ((threadIdx.x & 31) == 0) { red[w] = s; red[8 + w] = s2; }
    __syncthreads();
    float ts = 0.f, ts2 = 0.f;
    #pragma unroll
    for (int i = 0; i < 8; ++i) { ts += red[i]; ts2 += red[8 + i]; }
    mean = ts * (1.0f / (float)DIM);
    float var = ts2 * (1.0f / (float)DIM) - mean * mean;
    rstd = rsqrtf(var + LN_EPS);
}

// ---------------------------------------------------------------- elementwise kernels
__global__ void embed_kernel(const int* __restrict__ x,
                             const float* __restrict__ tok,
                             const float* __restrict__ pos,
                             float* __restrict__ h) {
    const int row = blockIdx.x;            // 0..4095  (b*SEQ + t)
    const int t   = threadIdx.x;           // 256 threads, 4 floats each
    const int id  = x[row];
    float4 a = reinterpret_cast<const float4*>(tok + (size_t)id * DIM)[t];
    float4 p = reinterpret_cast<const float4*>(pos + (size_t)(row % SEQ) * DIM)[t];
    a.x += p.x; a.y += p.y; a.z += p.z; a.w += p.w;
    reinterpret_cast<float4*>(h + (size_t)row * DIM)[t] = a;
}

// cos DFT matrix, bf16.  cos(2*pi*k*c/1024) -- period 1024, reduce (k*c) mod 1024.
__global__ void dft_cos_kernel(__bf16* __restrict__ cosm) {
    const int gid = blockIdx.x * 256 + threadIdx.x;   // 0 .. 1024*1024-1
    const int k = gid >> 10, c = gid & 1023;
    const int ph = (k * c) & 1023;
    cosm[gid] = (__bf16)cosf((float)ph * (6.283185307179586f / 1024.0f));
}

// Bulk f32 -> bf16 (weights staging; per-layer weights then live in L2).
__global__ void cvt_bf16_kernel(const float* __restrict__ src,
                                __bf16* __restrict__ dst) {
    const size_t i = ((size_t)blockIdx.x * 256 + threadIdx.x) * 4;
    float4 v = *reinterpret_cast<const float4*>(src + i);
    union { __bf16 h[4]; uint2 u; } o;
    o.h[0] = (__bf16)v.x; o.h[1] = (__bf16)v.y;
    o.h[2] = (__bf16)v.z; o.h[3] = (__bf16)v.w;
    *reinterpret_cast<uint2*>(dst + i) = o.u;
}

// dst = bf16( LN(src) * g + b )
__global__ void ln_to_bf16(const float* __restrict__ src,
                           const float* __restrict__ g,
                           const float* __restrict__ b,
                           __bf16* __restrict__ dst) {
    const int row = blockIdx.x, t = threadIdx.x;
    float4 x = reinterpret_cast<const float4*>(src + (size_t)row * DIM)[t];
    float mean, rstd; row_stats(x, mean, rstd);
    float4 gg = reinterpret_cast<const float4*>(g)[t];
    float4 bb = reinterpret_cast<const float4*>(b)[t];
    union { __bf16 h[4]; uint2 u; } o;
    o.h[0] = (__bf16)((x.x - mean) * rstd * gg.x + bb.x);
    o.h[1] = (__bf16)((x.y - mean) * rstd * gg.y + bb.y);
    o.h[2] = (__bf16)((x.z - mean) * rstd * gg.z + bb.z);
    o.h[3] = (__bf16)((x.w - mean) * rstd * gg.w + bb.w);
    reinterpret_cast<uint2*>(dst + (size_t)row * DIM)[t] = o.u;
}

// h += LN(f) * g + b   (residual after Fourier mixing)
__global__ void ln_res_kernel(const float* __restrict__ f,
                              const float* __restrict__ g,
                              const float* __restrict__ b,
                              float* __restrict__ h) {
    const int row = blockIdx.x, t = threadIdx.x;
    float4 x = reinterpret_cast<const float4*>(f + (size_t)row * DIM)[t];
    float mean, rstd; row_stats(x, mean, rstd);
    float4 gg = reinterpret_cast<const float4*>(g)[t];
    float4 bb = reinterpret_cast<const float4*>(b)[t];
    float4 hv = reinterpret_cast<const float4*>(h + (size_t)row * DIM)[t];
    hv.x += (x.x - mean) * rstd * gg.x + bb.x;
    hv.y += (x.y - mean) * rstd * gg.y + bb.y;
    hv.z += (x.z - mean) * rstd * gg.z + bb.z;
    hv.w += (x.w - mean) * rstd * gg.w + bb.w;
    reinterpret_cast<float4*>(h + (size_t)row * DIM)[t] = hv;
}

// ---------------------------------------------------------------- WMMA GEMM
// C[M,N] = A[M,K](bf16) * B[N,K]^T(bf16), K multiple of 64.
// Block: 256 threads = 8 waves as 4(M) x 2(N); wave tile 32(M) x 64(N).
// Block tile 128x128.  Grid: (N/128, M/128).
enum { EPI_STORE_F32 = 0, EPI_BIAS_GELU_BF16 = 1, EPI_BIAS_RES_F32 = 2 };

union Frag { u32x4 q[2]; v16bf v; };

__device__ __forceinline__ void load_a_frag(Frag& f, const __bf16* arow, int kb) {
    // ISA 16-bit A 16x32 layout: elems 0..7 -> K = kb+8*hi+(0..7),
    // elems 8..15 -> +16 (arow already includes the +8*hi lane-group offset).
    f.q[0] = *(GQ*)(arow + kb);
    f.q[1] = *(GQ*)(arow + kb + 16);
}
__device__ __forceinline__ void load_b_frag(Frag& f, const __bf16* brow, int kb) {
    // B 32x16 layout: elems 0..7 -> K = kb+8*hi+(0..7), elems 8..15 -> +16.
    f.q[0] = *(GQ*)(brow + kb);
    f.q[1] = *(GQ*)(brow + kb + 16);
}

__device__ __forceinline__ void load_set(Frag (&a)[2], Frag (&b)[4],
                                         const __bf16* const (&ap)[2],
                                         const __bf16* const (&bp)[4], int kofs) {
    #pragma unroll
    for (int mi = 0; mi < 2; ++mi) load_a_frag(a[mi], ap[mi], kofs);
    #pragma unroll
    for (int s = 0; s < 4; ++s)    load_b_frag(b[s], bp[s], kofs);
}

__device__ __forceinline__ void wmma_set(v8f (&acc)[2][4],
                                         const Frag (&a)[2], const Frag (&b)[4]) {
    #pragma unroll
    for (int mi = 0; mi < 2; ++mi)
        #pragma unroll
        for (int s = 0; s < 4; ++s)
            acc[mi][s] = __builtin_amdgcn_wmma_f32_16x16x32_bf16(
                false, a[mi].v, false, b[s].v, (short)0, acc[mi][s], false, false);
}

template<int EPI>
__global__ __launch_bounds__(256)
void gemm_wmma(const __bf16* __restrict__ A,
               const __bf16* __restrict__ Bw,
               const float*  __restrict__ bias,
               float*        __restrict__ outF,
               __bf16*       __restrict__ outB,
               int N, int K) {
    const int lane   = threadIdx.x & 31;
    const int wave   = threadIdx.x >> 5;
    const int wave_m = wave >> 1;          // 0..3
    const int wave_n = wave & 1;           // 0..1
    const int m0     = blockIdx.y * 128 + wave_m * 32;
    const int n0     = blockIdx.x * 128 + wave_n * 64;
    const int l16    = lane & 15;
    const int hi     = (lane >> 4) & 1;    // lane group

    v8f acc[2][4];
    #pragma unroll
    for (int mi = 0; mi < 2; ++mi)
        #pragma unroll
        for (int s = 0; s < 4; ++s)
            #pragma unroll
            for (int r = 0; r < 8; ++r) acc[mi][s][r] = 0.0f;

    // Persistent per-lane pointers, bumped by +64 elems per 64-K iteration.
    const __bf16* ap[2];
    #pragma unroll
    for (int mi = 0; mi < 2; ++mi)
        ap[mi] = A + (size_t)(m0 + mi * 16 + l16) * K + hi * 8;
    const __bf16* bp[4];
    #pragma unroll
    for (int s = 0; s < 4; ++s)
        bp[s] = Bw + (size_t)(n0 + s * 16 + l16) * K + hi * 8;

    // Double-buffered K-loop, step 64: set0 = even 32-blocks, set1 = odd.
    // All in-loop loads are unconditional (last 64 K peeled), so the compiler
    // can use partial s_wait_loadcnt instead of draining to zero.
    Frag a0[2], b0[4], a1[2], b1[4];
    load_set(a0, b0, ap, bp, 0);
    load_set(a1, b1, ap, bp, 32);

    for (int kb = 0; kb + 128 <= K; kb += 64) {
        wmma_set(acc, a0, b0);                 // consumes kb
        load_set(a0, b0, ap, bp, 64);          // prefetch kb+64
        wmma_set(acc, a1, b1);                 // consumes kb+32
        load_set(a1, b1, ap, bp, 96);          // prefetch kb+96
        #pragma unroll
        for (int mi = 0; mi < 2; ++mi) ap[mi] += 64;
        #pragma unroll
        for (int s = 0; s < 4; ++s)    bp[s]  += 64;
    }
    wmma_set(acc, a0, b0);                     // K-64
    wmma_set(acc, a1, b1);                     // K-32

    // Epilogue. C/D layout: VGPR r -> M = base + r + 8*hi, N = n0 + s*16 + l16.
    #pragma unroll
    for (int s = 0; s < 4; ++s) {
        const int n = n0 + s * 16 + l16;
        float bia = 0.0f;
        if (EPI != EPI_STORE_F32) bia = bias[n];
        #pragma unroll
        for (int mi = 0; mi < 2; ++mi) {
            #pragma unroll
            for (int r = 0; r < 8; ++r) {
                const int m = m0 + mi * 16 + r + hi * 8;
                const float v = acc[mi][s][r];
                if (EPI == EPI_STORE_F32) {
                    outF[(size_t)m * N + n] = v;
                } else if (EPI == EPI_BIAS_GELU_BF16) {
                    outB[(size_t)m * N + n] = (__bf16)gelu_exact(v + bia);
                } else { // EPI_BIAS_RES_F32
                    outF[(size_t)m * N + n] += v + bia;
                }
            }
        }
    }
}

// ---------------------------------------------------------------- driver
extern "C" void kernel_launch(void* const* d_in, const int* in_sizes, int n_in,
                              void* d_out, int out_size, void* d_ws, size_t ws_size,
                              hipStream_t stream) {
    const int*   X     = (const int*)  d_in[0];
    const float* tok   = (const float*)d_in[1];
    const float* pos   = (const float*)d_in[2];
    const float* ln1_g = (const float*)d_in[3];
    const float* ln1_b = (const float*)d_in[4];
    const float* lnF_g = (const float*)d_in[5];
    const float* lnF_b = (const float*)d_in[6];
    const float* ln2_g = (const float*)d_in[7];
    const float* ln2_b = (const float*)d_in[8];
    const float* W1    = (const float*)d_in[9];
    const float* b1    = (const float*)d_in[10];
    const float* W2    = (const float*)d_in[11];
    const float* b2    = (const float*)d_in[12];
    const float* lno_g = (const float*)d_in[13];
    const float* lno_b = (const float*)d_in[14];
    const float* headw = (const float*)d_in[15];
    float* logits = (float*)d_out;

    // Workspace carve (~110 MB).
    //  - shard holds (time-disjoint): f [f32 DFT out, 16.8MB], mid [bf16 GELU
    //    out, 33.5MB], head_bf [bf16 head weights, 65.5MB, used only after
    //    the layer loop].
    //  - w1b/w2b: per-layer bf16 weight staging (reused each layer).
    uint8_t* p = (uint8_t*)d_ws;
    auto carve = [&](size_t bytes) -> void* {
        void* r = (void*)p; p += (bytes + 255) & ~(size_t)255; return r;
    };
    float*  h      = (float*) carve((size_t)NTOK * DIM * 4);
    __bf16* abf    = (__bf16*)carve((size_t)NTOK * DIM * 2);
    void*   shard  = carve((size_t)VOCAB * DIM * 2);
    float*  f      = (float*) shard;
    __bf16* mid    = (__bf16*)shard;
    __bf16* headb  = (__bf16*)shard;
    __bf16* cosm   = (__bf16*)carve((size_t)DIM * DIM * 2);
    __bf16* w1b    = (__bf16*)carve((size_t)HID * DIM * 2);
    __bf16* w2b    = (__bf16*)carve((size_t)DIM * HID * 2);

    dft_cos_kernel<<<DIM * DIM / 256, 256, 0, stream>>>(cosm);
    embed_kernel<<<NTOK, 256, 0, stream>>>(X, tok, pos, h);

    for (int i = 0; i < DEPTH; ++i) {
        // a = LN1(h) ; f = real(fft(a)) = a @ Cos
        ln_to_bf16<<<NTOK, 256, 0, stream>>>(h, ln1_g + i * DIM, ln1_b + i * DIM, abf);
        gemm_wmma<EPI_STORE_F32>
            <<<dim3(DIM / 128, NTOK / 128), 256, 0, stream>>>(
                abf, cosm, nullptr, f, nullptr, DIM, DIM);
        // h += LN_F(f)
        ln_res_kernel<<<NTOK, 256, 0, stream>>>(f, lnF_g + i * DIM, lnF_b + i * DIM, h);
        // m = GELU(LN2(h) @ W1^T + b1)
        ln_to_bf16<<<NTOK, 256, 0, stream>>>(h, ln2_g + i * DIM, ln2_b + i * DIM, abf);
        cvt_bf16_kernel<<<HID * DIM / 1024, 256, 0, stream>>>(
            W1 + (size_t)i * HID * DIM, w1b);
        gemm_wmma<EPI_BIAS_GELU_BF16>
            <<<dim3(HID / 128, NTOK / 128), 256, 0, stream>>>(
                abf, w1b, b1 + (size_t)i * HID, nullptr, mid, HID, DIM);
        // h += m @ W2^T + b2
        cvt_bf16_kernel<<<DIM * HID / 1024, 256, 0, stream>>>(
            W2 + (size_t)i * DIM * HID, w2b);
        gemm_wmma<EPI_BIAS_RES_F32>
            <<<dim3(DIM / 128, NTOK / 128), 256, 0, stream>>>(
                mid, w2b, b2 + (size_t)i * DIM, h, nullptr, DIM, HID);
    }

    // logits = LN_out(h) @ head_w^T   (mid/f dead -> shard reused for bf16 head)
    ln_to_bf16<<<NTOK, 256, 0, stream>>>(h, lno_g, lno_b, abf);
    cvt_bf16_kernel<<<VOCAB * DIM / 1024, 256, 0, stream>>>(headw, headb);
    gemm_wmma<EPI_STORE_F32>
        <<<dim3(VOCAB / 128, NTOK / 128), 256, 0, stream>>>(
            abf, headb, nullptr, logits, nullptr, VOCAB, DIM);
}
